// PPFNet_317827580545
// MI455X (gfx1250) — compile-verified, matrix operand order checked by reading, not run on hardware
//
#include <hip/hip_runtime.h>
#include <hip/hip_bf16.h>
#include <stdint.h>

#define RADIUS2  0.09f
#define NS       64
#define NPTS     2048
#define NBATCH   8
#define GN_EPS   1e-5f

typedef __attribute__((ext_vector_type(16))) _Float16 v16h;
typedef __attribute__((ext_vector_type(8)))  float    v8f;

struct alignas(16) U4 { unsigned int x, y, z, w; };
union HFrag { v16h v; U4 q[2]; _Float16 h[16]; };
union H8 { U4 q; _Float16 h[8]; };

__device__ __forceinline__ unsigned int pack2h(float a, float b) {
  union { _Float16 h[2]; unsigned int u; } z;
  z.h[0] = (_Float16)a; z.h[1] = (_Float16)b;
  return z.u;
}

// ---------------------------------------------------------------------------
// Weight pack: f32 [O][I] -> f16 [O][Kp] zero-padded (Kp multiple of 32)
// ---------------------------------------------------------------------------
__global__ void pack_w(const float* __restrict__ w, _Float16* __restrict__ o,
                       int O, int I, int Kp) {
  int i = blockIdx.x * blockDim.x + threadIdx.x;
  if (i >= O * Kp) return;
  int r = i / Kp, k = i - r * Kp;
  o[i] = (k < I) ? (_Float16)w[r * I + k] : (_Float16)0.f;
}

__global__ void zero_f(float* __restrict__ p, int n) {
  int i = blockIdx.x * blockDim.x + threadIdx.x;
  if (i < n) p[i] = 0.f;
}

// ---------------------------------------------------------------------------
// Ball query + PPF features. One wave32 per point, ordered ballot-append of
// in-radius indices (== reference sort-by-index, first NSAMPLE, backfill with
// first neighbor). Writes fused [x(3), d(3), ppf(4)] padded to 32 ch f16,
// layout [S0][32], S0 ordered (b, n, sample).
// ---------------------------------------------------------------------------
__device__ __forceinline__ float angle3(float ax, float ay, float az,
                                        float bx, float by, float bz) {
  float cx = ay * bz - az * by;
  float cy = az * bx - ax * bz;
  float cz = ax * by - ay * bx;
  float cn = cx * cx + cy * cy + cz * cz;
  float dot = ax * bx + ay * by + az * bz;
  bool deg = (cn < 1e-20f) && (fabsf(dot) < 1e-12f);
  float cns = (cn < 1e-20f) ? 0.f : sqrtf(cn);
  float ang = atan2f(cns, deg ? 1.0f : dot);
  return deg ? 0.f : ang;
}

__global__ void __launch_bounds__(256)
group_ppf(const float* __restrict__ xyz, const float* __restrict__ nrm,
          _Float16* __restrict__ F0) {
  __shared__ int nbr[8][NS];
  __shared__ int cnt[8];
  const int w = threadIdx.x >> 5, lane = threadIdx.x & 31;
  const int p = blockIdx.x * 8 + w;           // global point id = b*N + n
  const int b = p >> 11, n = p & (NPTS - 1);
  const float* xb = xyz + (size_t)b * NPTS * 3;
  const float* nb_ = nrm + (size_t)b * NPTS * 3;
  const float cx = xb[n * 3 + 0], cy = xb[n * 3 + 1], cz = xb[n * 3 + 2];
  const float nx = nb_[n * 3 + 0], ny = nb_[n * 3 + 1], nz = nb_[n * 3 + 2];
  if (lane == 0) cnt[w] = 0;
  __syncthreads();
  for (int j0 = 0; j0 < NPTS; j0 += 32) {
    int j = j0 + lane;
    float dx = xb[j * 3 + 0] - cx, dy = xb[j * 3 + 1] - cy, dz = xb[j * 3 + 2] - cz;
    bool ok = (dx * dx + dy * dy + dz * dz) <= RADIUS2;
    unsigned mask = __builtin_amdgcn_ballot_w32(ok);
    int c = cnt[w];
    int pos = c + __popc(mask & ((1u << lane) - 1u));
    if (ok && pos < NS) nbr[w][pos] = j;
    __syncthreads();
    if (lane == 0) cnt[w] = min(NS, c + __popc(mask));
    __syncthreads();
  }
  int c = cnt[w];
  int first = nbr[w][0];   // self always qualifies -> c >= 1
  for (int q = lane; q < NS; q += 32)
    if (q >= c) nbr[w][q] = first;
  __syncthreads();
#pragma unroll
  for (int t = 0; t < 2; t++) {
    int q = lane + 32 * t;
    int idx = nbr[w][q];
    float gx = xb[idx * 3 + 0], gy = xb[idx * 3 + 1], gz = xb[idx * 3 + 2];
    float gnx = nb_[idx * 3 + 0], gny = nb_[idx * 3 + 1], gnz = nb_[idx * 3 + 2];
    float dx = gx - cx, dy = gy - cy, dz = gz - cz;
    float a0 = angle3(nx, ny, nz, dx, dy, dz);
    float a1 = angle3(gnx, gny, gnz, dx, dy, dz);
    float a2 = angle3(nx, ny, nz, gnx, gny, gnz);
    float ss = dx * dx + dy * dy + dz * dz;
    float nm = (ss < 1e-20f) ? 0.f : sqrtf(ss);
    _Float16* row = F0 + ((size_t)p * NS + q) * 32;
    row[0] = (_Float16)cx; row[1] = (_Float16)cy; row[2] = (_Float16)cz;
    row[3] = (_Float16)dx; row[4] = (_Float16)dy; row[5] = (_Float16)dz;
    row[6] = (_Float16)a0; row[7] = (_Float16)a1; row[8] = (_Float16)a2;
    row[9] = (_Float16)nm;
#pragma unroll
    for (int cc = 10; cc < 32; cc++) row[cc] = (_Float16)0.f;
  }
}

// ---------------------------------------------------------------------------
// WMMA GEMM: Y[s][c] = sum_k W[c][k] * X[s][k] + bias[c]
//   W: f16 [Cout][K] row-major (K mult of 32), X: f16 [S][K], Y: f16 [S][Cout]
// One wave: 32 channels (2 M-tiles) x 64 columns (4 N-tiles), K-loop of 32.
// M-block index is the FAST wid index so all waves covering the same columns
// run adjacently -> the streamed X tile stays hot in WGP$/L2 and HBM sees it
// once. Each B fragment feeds 2 WMMAs; 8 WMMAs per 12 b128 loads.
// ---------------------------------------------------------------------------
__global__ void __launch_bounds__(256)
gemm_bias(const _Float16* __restrict__ W, const float* __restrict__ bias,
          const _Float16* __restrict__ X, _Float16* __restrict__ Y,
          int Cout, int K, int S) {
  const int lane = threadIdx.x & 31;
  const int hi = lane >> 4;
  const int lo16 = lane & 15;
  const int mt2 = Cout >> 5;                  // # of 32-channel blocks
  int wid = blockIdx.x * 8 + (threadIdx.x >> 5);
  int nb = wid / mt2;                         // 64-column block (slow)
  int mb = wid - nb * mt2;                    // 32-channel block (fast)
  if (nb >= (S >> 6)) return;
  const int m0 = mb << 5;
  const int s0 = nb << 6;

  v8f acc[2][4] = {};
  const _Float16* wrow0 = W + (size_t)(m0 + lo16) * K + 8 * hi;
  const _Float16* wrow1 = wrow0 + (size_t)16 * K;
  const _Float16* xcol0 = X + (size_t)(s0 + lo16) * K + 8 * hi;
  const size_t cs = (size_t)16 * K;

  for (int k0 = 0; k0 < K; k0 += 32) {
    HFrag a0, a1;
    a0.q[0] = *(const U4*)(wrow0 + k0);
    a0.q[1] = *(const U4*)(wrow0 + k0 + 16);
    a1.q[0] = *(const U4*)(wrow1 + k0);
    a1.q[1] = *(const U4*)(wrow1 + k0 + 16);
    __builtin_prefetch(xcol0 + k0 + 32, 0, 1);          // global_prefetch_b8
    __builtin_prefetch(xcol0 + 3 * cs + k0 + 32, 0, 1);
#pragma unroll
    for (int t = 0; t < 4; t++) {
      HFrag b;
      b.q[0] = *(const U4*)(xcol0 + t * cs + k0);
      b.q[1] = *(const U4*)(xcol0 + t * cs + k0 + 16);
      acc[0][t] = __builtin_amdgcn_wmma_f32_16x16x32_f16(
          false, a0.v, false, b.v, (short)0, acc[0][t], false, false);
      acc[1][t] = __builtin_amdgcn_wmma_f32_16x16x32_f16(
          false, a1.v, false, b.v, (short)0, acc[1][t], false, false);
    }
  }

  // Epilogue: acc VGPR r of (u,t) -> channel m0 + 16u + 8*hi + r (8 contiguous
  // channels), column s0 + 16t + lo16 -> one b128 store each.
#pragma unroll
  for (int u = 0; u < 2; u++) {
    const int cbase = m0 + 16 * u + 8 * hi;
    float bs[8];
#pragma unroll
    for (int r = 0; r < 8; r++) bs[r] = bias[cbase + r];
#pragma unroll
    for (int t = 0; t < 4; t++) {
      int col = s0 + 16 * t + lo16;
      v8f a = acc[u][t];
      U4 o;
      o.x = pack2h(a[0] + bs[0], a[1] + bs[1]);
      o.y = pack2h(a[2] + bs[2], a[3] + bs[3]);
      o.z = pack2h(a[4] + bs[4], a[5] + bs[5]);
      o.w = pack2h(a[6] + bs[6], a[7] + bs[7]);
      *(U4*)(Y + (size_t)col * Cout + cbase) = o;
    }
  }
}

// ---------------------------------------------------------------------------
// GroupNorm pass 1: one thread accumulates a whole [C] row into 8 per-group
// register partials, then 16 LDS atomics per row (vs 2 per element before).
// stats layout: [b*8+g] -> {sum, sumsq}  (128 floats, B=8, G=8)
// ---------------------------------------------------------------------------
__global__ void __launch_bounds__(256)
gn_reduce(const _Float16* __restrict__ Y, float* __restrict__ st,
          int C, int SperB, int S) {
  __shared__ float ls[128];
  for (int i = threadIdx.x; i < 128; i += blockDim.x) ls[i] = 0.f;
  __syncthreads();
  int cpg = C >> 3;
  for (int s = blockIdx.x * blockDim.x + threadIdx.x; s < S;
       s += gridDim.x * blockDim.x) {
    int b = s / SperB;
    const _Float16* row = Y + (size_t)s * C;
    float s1[8], s2[8];
#pragma unroll
    for (int g = 0; g < 8; g++) { s1[g] = 0.f; s2[g] = 0.f; }
    for (int g = 0; g < 8; g++)
      for (int k = 0; k < cpg; k++) {
        float v = (float)row[g * cpg + k];
        s1[g] += v; s2[g] += v * v;
      }
#pragma unroll
    for (int g = 0; g < 8; g++) {
      atomicAdd(&ls[(b * 8 + g) * 2], s1[g]);       // ds_add_f32
      atomicAdd(&ls[(b * 8 + g) * 2 + 1], s2[g]);
    }
  }
  __syncthreads();
  for (int i = threadIdx.x; i < 128; i += blockDim.x)
    if (ls[i] != 0.f) atomicAdd(&st[i], ls[i]);
}

// GroupNorm pass 2: in-place normalize + affine + ReLU, b128-vectorized.
__global__ void __launch_bounds__(256)
gn_norm_relu(_Float16* __restrict__ Y, const float* __restrict__ st,
             const float* __restrict__ gamma, const float* __restrict__ beta,
             int C, int SperB, size_t nChunks) {
  size_t i = (size_t)blockIdx.x * blockDim.x + threadIdx.x;
  if (i >= nChunks) return;
  size_t base = i * 8;
  size_t s = base / C;
  int c0 = (int)(base - s * C);
  int b = (int)(s / SperB);
  int cpg = C >> 3;
  float cnt = (float)cpg * (float)SperB;
  H8 d; d.q = *(const U4*)(Y + base);
  H8 o;
#pragma unroll
  for (int j = 0; j < 8; j++) {
    int c = c0 + j;
    int g = c / cpg;
    float m = st[(b * 8 + g) * 2] / cnt;
    float var = st[(b * 8 + g) * 2 + 1] / cnt - m * m;
    float inv = rsqrtf(var + GN_EPS);
    float v = ((float)d.h[j] - m) * inv * gamma[c] + beta[c];
    o.h[j] = (_Float16)fmaxf(v, 0.f);
  }
  *(U4*)(Y + base) = o.q;
}

// Max-pool over 64 samples per point, 8 channels (b128) per thread.
__global__ void __launch_bounds__(256)
maxpool64(const _Float16* __restrict__ X, _Float16* __restrict__ P, int C) {
  int chunks = C >> 3;
  int i = blockIdx.x * blockDim.x + threadIdx.x;
  int tot = NBATCH * NPTS * chunks;
  if (i >= tot) return;
  int cc = i % chunks;
  int pt = i / chunks;
  const _Float16* base = X + (size_t)pt * NS * C + cc * 8;
  float m[8];
#pragma unroll
  for (int e = 0; e < 8; e++) m[e] = -1e30f;
  for (int j = 0; j < NS; j++) {
    H8 d; d.q = *(const U4*)(base + (size_t)j * C);
#pragma unroll
    for (int e = 0; e < 8; e++) m[e] = fmaxf(m[e], (float)d.h[e]);
  }
  H8 o;
#pragma unroll
  for (int e = 0; e < 8; e++) o.h[e] = (_Float16)m[e];
  *(U4*)(P + (size_t)pt * C + cc * 8) = o.q;
}

// Final: L2-normalize 96 channels/point (wave32 butterfly: 32 lanes x 3 ch),
// write f32 output [B][N][96].
__global__ void __launch_bounds__(256)
l2norm_out(const _Float16* __restrict__ Yf, float* __restrict__ out) {
  const int w = threadIdx.x >> 5, lane = threadIdx.x & 31;
  const int p = blockIdx.x * 8 + w;
  const _Float16* row = Yf + (size_t)p * 96;
  float v0 = (float)row[lane * 3 + 0];
  float v1 = (float)row[lane * 3 + 1];
  float v2 = (float)row[lane * 3 + 2];
  float ss = v0 * v0 + v1 * v1 + v2 * v2;
#pragma unroll
  for (int m = 16; m >= 1; m >>= 1) ss += __shfl_xor(ss, m, 32);
  float inv = rsqrtf(fmaxf(ss, 1e-20f));
  float* orow = out + (size_t)p * 96;
  orow[lane * 3 + 0] = v0 * inv;
  orow[lane * 3 + 1] = v1 * inv;
  orow[lane * 3 + 2] = v2 * inv;
}

// ---------------------------------------------------------------------------
extern "C" void kernel_launch(void* const* d_in, const int* in_sizes, int n_in,
                              void* d_out, int out_size, void* d_ws, size_t ws_size,
                              hipStream_t stream) {
  const float* xyz     = (const float*)d_in[0];
  const float* nrm     = (const float*)d_in[1];
  const float* pre_w1  = (const float*)d_in[2];
  const float* pre_b1  = (const float*)d_in[3];
  const float* pre_g1  = (const float*)d_in[4];
  const float* pre_be1 = (const float*)d_in[5];
  const float* pre_w2  = (const float*)d_in[6];
  const float* pre_b2  = (const float*)d_in[7];
  const float* pre_g2  = (const float*)d_in[8];
  const float* pre_be2 = (const float*)d_in[9];
  const float* pre_w3  = (const float*)d_in[10];
  const float* pre_b3  = (const float*)d_in[11];
  const float* pre_g3  = (const float*)d_in[12];
  const float* pre_be3 = (const float*)d_in[13];
  const float* post_w1 = (const float*)d_in[14];
  const float* post_b1 = (const float*)d_in[15];
  const float* post_g1 = (const float*)d_in[16];
  const float* post_be1= (const float*)d_in[17];
  const float* post_w2 = (const float*)d_in[18];
  const float* post_b2 = (const float*)d_in[19];
  const float* post_g2 = (const float*)d_in[20];
  const float* post_be2= (const float*)d_in[21];
  const float* post_w3 = (const float*)d_in[22];
  const float* post_b3 = (const float*)d_in[23];

  // Workspace layout (bytes). bufA >= 192MB, bufB >= 384MB (ping-pong; GN
  // normalize runs in place so each layer needs only in+out buffers).
  char* ws = (char*)d_ws;
  _Float16* bufA  = (_Float16*)(ws);                    // F0 / Y2
  _Float16* bufB  = (_Float16*)(ws + 201326592ull);     // Y1 / Y3
  _Float16* pool  = (_Float16*)(ws + 603979776ull);     // [16384][192]
  _Float16* bufD  = (_Float16*)(ws + 610271232ull);     // post1 out
  _Float16* bufE  = (_Float16*)(ws + 616562688ull);     // post2 out
  _Float16* bufF  = (_Float16*)(ws + 619708416ull);     // post3 out
  float*    stats = (float*)  (ws + 622854144ull);      // 128 floats
  _Float16* w1p = (_Float16*)(ws + 622854656ull);       // 96x32
  _Float16* w2p = w1p + 96 * 32;                        // 96x96
  _Float16* w3p = w2p + 96 * 96;                        // 192x96
  _Float16* pw1 = w3p + 192 * 96;                       // 192x192
  _Float16* pw2 = pw1 + 192 * 192;                      // 96x192
  _Float16* pw3 = pw2 + 96 * 192;                       // 96x96

  auto pack = [&](const float* w, _Float16* o, int O, int I, int Kp) {
    int n = O * Kp;
    pack_w<<<(n + 255) / 256, 256, 0, stream>>>(w, o, O, I, Kp);
  };
  pack(pre_w1, w1p, 96, 10, 32);
  pack(pre_w2, w2p, 96, 96, 96);
  pack(pre_w3, w3p, 192, 96, 96);
  pack(post_w1, pw1, 192, 192, 192);
  pack(post_w2, pw2, 96, 192, 192);
  pack(post_w3, pw3, 96, 96, 96);

  group_ppf<<<(NBATCH * NPTS) / 8, 256, 0, stream>>>(xyz, nrm, bufA);

  auto gemm = [&](const _Float16* W, const float* bias, const _Float16* X,
                  _Float16* Y, int Cout, int K, int S) {
    int waves = (Cout >> 5) * (S >> 6);
    gemm_bias<<<(waves + 7) / 8, 256, 0, stream>>>(W, bias, X, Y, Cout, K, S);
  };
  auto gnorm = [&](_Float16* Y, const float* g, const float* be, int C,
                   int SperB, int S) {
    zero_f<<<1, 128, 0, stream>>>(stats, 128);
    gn_reduce<<<2048, 256, 0, stream>>>(Y, stats, C, SperB, S);
    size_t nChunks = ((size_t)S * C) / 8;
    gn_norm_relu<<<(unsigned)((nChunks + 255) / 256), 256, 0, stream>>>(
        Y, stats, g, be, C, SperB, nChunks);
  };

  const int S0 = NBATCH * NPTS * NS;  // 1,048,576
  const int S1 = NBATCH * NPTS;       // 16,384

  gemm(w1p, pre_b1, bufA, bufB, 96, 32, S0);
  gnorm(bufB, pre_g1, pre_be1, 96, NPTS * NS, S0);
  gemm(w2p, pre_b2, bufB, bufA, 96, 96, S0);
  gnorm(bufA, pre_g2, pre_be2, 96, NPTS * NS, S0);
  gemm(w3p, pre_b3, bufA, bufB, 192, 96, S0);
  gnorm(bufB, pre_g3, pre_be3, 192, NPTS * NS, S0);

  maxpool64<<<(S1 * (192 / 8) + 255) / 256, 256, 0, stream>>>(bufB, pool, 192);

  gemm(pw1, post_b1, pool, bufD, 192, 192, S1);
  gnorm(bufD, post_g1, post_be1, 192, NPTS, S1);
  gemm(pw2, post_b2, bufD, bufE, 96, 192, S1);
  gnorm(bufE, post_g2, post_be2, 96, NPTS, S1);
  gemm(pw3, post_b3, bufE, bufF, 96, 96, S1);

  l2norm_out<<<S1 / 8, 256, 0, stream>>>(bufF, (float*)d_out);
}
